// LDALoss_33621003993451
// MI455X (gfx1250) — matrix operand order repaired and checked by reading, not compiled, not run on hardware
//
#include <hip/hip_runtime.h>
#include <math.h>

// Problem constants (fixed by the reference: path_fea (131072, 64, 1, 1, 2) f32)
#define NUM_B 131072
#define NUM_D 128          // 64 blocks * 2
#define NUM_P 16           // samples per group
#define NUM_G (NUM_B / NUM_P)   // 8192 centers
#define GT (NUM_G / 16)    // 512 WMMA tile rows/cols

typedef __attribute__((ext_vector_type(16))) _Float16 v16h;
typedef __attribute__((ext_vector_type(8)))  _Float16 v8h;
typedef __attribute__((ext_vector_type(4)))  _Float16 v4h;
typedef __attribute__((ext_vector_type(8)))  float    v8f;

// Raw v_sqrt_f32 (fast HW sqrt, no Newton refinement — ample for a hinge loss)
__device__ __forceinline__ float fast_sqrt(float x) {
    return __builtin_amdgcn_sqrtf(x);
}

__device__ __forceinline__ float waveReduceSum(float v) {
#pragma unroll
    for (int off = 16; off > 0; off >>= 1)
        v += __shfl_xor(v, off, 32);
    return v;
}

// ---------------------------------------------------------------------------
// Kernel 1: one wave per group.
//  - compute center (mean of 16 rows), store as f16 row-major [G][D]
//  - compute ||center||^2 -> sq[g]
//  - compute intra hinge sum for the group's 16 samples
// Lane l owns columns 4l..4l+3 -> perfectly coalesced b128 loads (512B/row).
// ---------------------------------------------------------------------------
__global__ __launch_bounds__(256)
void k_center_intra(const float* __restrict__ fea,
                    _Float16* __restrict__ center,
                    float* __restrict__ sq,
                    float* __restrict__ partials)
{
    const int wave = __builtin_amdgcn_readfirstlane(threadIdx.x >> 5);
    const int lane = threadIdx.x & 31;
    const int g    = blockIdx.x * 8 + wave;

    const float4* base = (const float4*)(fea + (size_t)g * NUM_P * NUM_D);
    float4 rows[NUM_P];
    float sx = 0.f, sy = 0.f, sz = 0.f, sw = 0.f;
#pragma unroll
    for (int r = 0; r < NUM_P; ++r) {
        rows[r] = base[r * (NUM_D / 4) + lane];
        sx += rows[r].x; sy += rows[r].y; sz += rows[r].z; sw += rows[r].w;
    }
    const float inv = 1.0f / (float)NUM_P;
    const float cx = sx * inv, cy = sy * inv, cz = sz * inv, cw = sw * inv;

    // squared norm of the center (cross-lane reduce over all 128 cols)
    const float nrm = waveReduceSum(cx * cx + cy * cy + cz * cz + cw * cw);
    if (lane == 0) sq[g] = nrm;

    // store center as f16 (8-byte store per lane)
    v4h cv;
    cv[0] = (_Float16)cx; cv[1] = (_Float16)cy;
    cv[2] = (_Float16)cz; cv[3] = (_Float16)cw;
    *(v4h*)(center + (size_t)g * NUM_D + 4 * lane) = cv;

    // intra hinge: per sample distance to center
    float acc = 0.f;
#pragma unroll
    for (int r = 0; r < NUM_P; ++r) {
        const float dx = rows[r].x - cx, dy = rows[r].y - cy;
        const float dz = rows[r].z - cz, dw = rows[r].w - cw;
        const float t = waveReduceSum(dx * dx + dy * dy + dz * dz + dw * dw);
        if (lane == 0) {
            const float d = fast_sqrt(t);               // v_sqrt_f32
            const float h = fmaxf(d - 0.1f, 0.f);       // MARGIN_INTRA
            acc = fmaf(h, h, acc);
        }
    }

    __shared__ float red[8];
    if (lane == 0) red[wave] = acc;
    __syncthreads();
    if (threadIdx.x == 0) {
        float s = 0.f;
#pragma unroll
        for (int i = 0; i < 8; ++i) s += red[i];
        partials[blockIdx.x] = s;
    }
}

// ---------------------------------------------------------------------------
// Kernel 2: inter loss via fused WMMA Gram + hinge reduction.
// Grid: (16, 512). Each wave handles a 16x64 strip: tile row ti (blockIdx.y),
// 4 consecutive tile cols starting at (blockIdx.x*8 + wave)*4.
// Gram tile = sum_k A(16x32) * B(32x16) via v_wmma_f32_16x16x32_f16 (4 steps).
// Upper triangle only. All branches around WMMA are *scalar* (readfirstlane-
// forced uniformity) so EXEC stays all-1s as WMMA requires; the per-element
// triangle mask is a branchless select.
// ---------------------------------------------------------------------------
__global__ __launch_bounds__(256)
void k_inter(const _Float16* __restrict__ center,
             const float* __restrict__ sq,
             float* __restrict__ partials)
{
    const int wave = __builtin_amdgcn_readfirstlane(threadIdx.x >> 5);
    const int lane = threadIdx.x & 31;
    const int lm = lane & 15;
    const int hi = lane >> 4;

    const int ti     = blockIdx.y;
    const int tjBase = (blockIdx.x * 8 + wave) * 4;

    float acc = 0.f;

    if (tjBase + 3 >= ti) {            // strip intersects upper triangle (scalar)
        const int aRow = ti * 16 + lm;
        const _Float16* aptr = center + (size_t)aRow * NUM_D;

        // A fragments (16x32 f16 layout, ISA 7.12.2): two 16B chunks per step
        union { v16h v; v8h h[2]; } afrag[4];
#pragma unroll
        for (int s = 0; s < 4; ++s) {
            const int kk = s * 32;
            afrag[s].h[0] = *(const v8h*)(aptr + kk + hi * 8);
            afrag[s].h[1] = *(const v8h*)(aptr + kk + 16 + hi * 8);
        }

        // sq[i] for this lane's 8 output rows (i = ti*16 + hi*8 + v)
        const float4* sp = (const float4*)(sq + ti * 16 + hi * 8);
        const float4 s0 = sp[0], s1 = sp[1];
        const float sqi[8] = { s0.x, s0.y, s0.z, s0.w, s1.x, s1.y, s1.z, s1.w };
        const int iBase = ti * 16 + hi * 8;

#pragma unroll
        for (int t = 0; t < 4; ++t) {
            const int tj = tjBase + t;
            if (tj < ti) continue;     // scalar branch (tj, ti uniform)
            const int bRow = tj * 16 + lm;
            const _Float16* bptr = center + (size_t)bRow * NUM_D + hi * 8;

            v8f c = {};
#pragma unroll
            for (int s = 0; s < 4; ++s) {
                union { v16h v; v8h h[2]; } bu;
                bu.h[0] = *(const v8h*)(bptr + s * 32);        // 16B aligned loads
                bu.h[1] = *(const v8h*)(bptr + s * 32 + 16);
                const v16h b = bu.v;
                c = __builtin_amdgcn_wmma_f32_16x16x32_f16(
                        false, afrag[s].v, false, b, (short)0, c, false, false);
            }

            const int j = tj * 16 + lm;
            const float sqj = sq[j];
#pragma unroll
            for (int v = 0; v < 8; ++v) {
                float d2 = fmaxf(sqi[v] + sqj - 2.0f * c[v], 0.f);
                float d  = fast_sqrt(d2);                    // v_sqrt_f32
                float h  = fmaxf(1.0f - d, 0.f);             // MARGIN_INTER
                h = (j > iBase + v) ? h : 0.f;               // branchless tri mask
                acc = fmaf(h, h, acc);
            }
        }
    }

    const float w = waveReduceSum(acc);
    __shared__ float red[8];
    if (lane == 0) red[wave] = w;
    __syncthreads();
    if (threadIdx.x == 0) {
        float s = 0.f;
#pragma unroll
        for (int i = 0; i < 8; ++i) s += red[i];
        partials[blockIdx.y * gridDim.x + blockIdx.x] = s;
    }
}

// ---------------------------------------------------------------------------
// Kernel 3: final reduction of partials -> (inter_loss, intra_loss)
// ---------------------------------------------------------------------------
__global__ __launch_bounds__(256)
void k_finalize(const float* __restrict__ pInter, int nInter,
                const float* __restrict__ pIntra, int nIntra,
                float* __restrict__ out)
{
    __shared__ float s1[256], s2[256];
    float a = 0.f, b = 0.f;
    for (int i = threadIdx.x; i < nInter; i += 256) a += pInter[i];
    for (int i = threadIdx.x; i < nIntra; i += 256) b += pIntra[i];
    s1[threadIdx.x] = a; s2[threadIdx.x] = b;
    __syncthreads();
    for (int st = 128; st > 0; st >>= 1) {
        if (threadIdx.x < st) {
            s1[threadIdx.x] += s1[threadIdx.x + st];
            s2[threadIdx.x] += s2[threadIdx.x + st];
        }
        __syncthreads();
    }
    if (threadIdx.x == 0) {
        const float nPairs = 0.5f * (float)NUM_G * (float)(NUM_G - 1);
        out[0] = s1[0] / nPairs;                         // inter_loss
        out[1] = s2[0] / (float)(NUM_G * NUM_P);         // intra_loss
    }
}

extern "C" void kernel_launch(void* const* d_in, const int* in_sizes, int n_in,
                              void* d_out, int out_size, void* d_ws, size_t ws_size,
                              hipStream_t stream)
{
    (void)in_sizes; (void)n_in; (void)out_size; (void)ws_size;
    const float* fea = (const float*)d_in[0];
    float* out = (float*)d_out;

    // Workspace layout (~2.17 MB total):
    //   center f16 [G][D]            : 2 MiB
    //   sq [G] f32                   : 32 KiB
    //   intra partials (1024 f32)    : 4 KiB
    //   inter partials (8192 f32)    : 32 KiB
    char* ws = (char*)d_ws;
    _Float16* center = (_Float16*)ws;
    float* sq     = (float*)(ws + (size_t)NUM_G * NUM_D * sizeof(_Float16));
    float* pIntra = sq + NUM_G;
    float* pInter = pIntra + (NUM_G / 8);

    // Phase 1: centers + intra loss (memory-bound streaming, 64 MiB read)
    k_center_intra<<<dim3(NUM_G / 8), dim3(256), 0, stream>>>(fea, center, sq, pIntra);

    // Phase 2: inter loss — WMMA Gram over the 2 MiB f16 center matrix (L2-resident)
    k_inter<<<dim3(16, GT), dim3(256), 0, stream>>>(center, sq, pInter);

    // Phase 3: finalize both scalars
    k_finalize<<<dim3(1), dim3(256), 0, stream>>>(pInter, 16 * GT, pIntra, NUM_G / 8, out);
}
